// baseFreeSolvModel_26611617366460
// MI455X (gfx1250) — matrix operand order repaired and checked by reading, not compiled, not run on hardware
//
#include <hip/hip_runtime.h>
#include <hip/hip_bf16.h>
#include <math.h>

#define N_NODES  50000
#define N_EDGES  800000
#define N_FEAT   32
#define EMB      64
#define N_GRAPHS 512

typedef __attribute__((ext_vector_type(16))) _Float16 v16h;
typedef __attribute__((ext_vector_type(8)))  _Float16 v8h;
typedef __attribute__((ext_vector_type(8)))  float    v8f;

// ---------------------------------------------------------------- utilities
__global__ void k_fill(float* __restrict__ p, int n, float v) {
  int i = blockIdx.x * blockDim.x + threadIdx.x;
  if (i < n) p[i] = v;
}

// deg[i] starts at 1.0 (self loop); add 1 per incoming edge
__global__ void k_deg_edges(const int* __restrict__ dst, float* __restrict__ deg) {
  int e = blockIdx.x * blockDim.x + threadIdx.x;
  if (e < N_EDGES) atomicAdd(&deg[dst[e]], 1.0f);
}

// dis = rsqrt(deg); selfnorm = dis^2 = 1/deg
__global__ void k_deg_fin(float* __restrict__ dis, float* __restrict__ selfn) {
  int i = blockIdx.x * blockDim.x + threadIdx.x;
  if (i < N_NODES) {
    float r = rsqrtf(dis[i]);
    dis[i]   = r;
    selfn[i] = r * r;
  }
}

__global__ void k_edge_norm(const int* __restrict__ src, const int* __restrict__ dst,
                            const float* __restrict__ dis, float* __restrict__ norm) {
  int e = blockIdx.x * blockDim.x + threadIdx.x;
  if (e < N_EDGES) norm[e] = dis[src[e]] * dis[dst[e]];
}

// ------------------------------------------------------- dense matmul (WMMA)
// Hout[50000 x 64] = Hin[50000 x K] @ W[K x 64], K in {32, 64} (compile-time).
// One block = 16 node rows (3125 * 16 = 50000 exactly); 4 waves, each owns a
// 16-wide N tile. W is staged TRANSPOSED into LDS as f16 with padded stride so
// each lane's B fragment is two 8-half runs (k +0..7, +16..23) -> 2x ds_load_b128.
#define WSTRIDE 72   // 64 + 8 halves: kills bank conflicts, keeps 16B alignment

template <int K>
__global__ __launch_bounds__(128) void gcn_matmul_wmma(const float* __restrict__ Hin,
                                                       const float* __restrict__ W,
                                                       float* __restrict__ Hout) {
  __shared__ _Float16 WldsT[EMB * WSTRIDE];   // [n][k], padded
  const int tid = threadIdx.x;
  for (int idx = tid; idx < K * EMB; idx += 128) {
    int k = idx >> 6, n = idx & 63;           // W is [K][64] row-major
    WldsT[n * WSTRIDE + k] = (_Float16)W[idx];
  }
  __syncthreads();

  const int wave = tid >> 5;            // 0..3 -> N tile
  const int lane = tid & 31;
  const int m    = lane & 15;           // row within tile / col within B
  const int hi   = lane >> 4;           // half-wave selector
  const int row0 = blockIdx.x * 16;
  const int n0   = wave * 16;

  v8f acc = {};
#pragma unroll
  for (int k0 = 0; k0 < K; k0 += 32) {
    // A fragment (16x32 f16): elems 0..7  -> K = k0 + hi*8 + j
    //                         elems 8..15 -> K = k0 + hi*8 + 16 + j
    const float* arow = Hin + (size_t)(row0 + m) * K + k0 + hi * 8;  // 32B aligned
    const float4* a4 = (const float4*)arow;
    float4 f0 = a4[0], f1 = a4[1];      // K offsets +0..7
    float4 f2 = a4[4], f3 = a4[5];      // K offsets +16..23
    v16h a;
    a[0]  = (_Float16)f0.x; a[1]  = (_Float16)f0.y; a[2]  = (_Float16)f0.z; a[3]  = (_Float16)f0.w;
    a[4]  = (_Float16)f1.x; a[5]  = (_Float16)f1.y; a[6]  = (_Float16)f1.z; a[7]  = (_Float16)f1.w;
    a[8]  = (_Float16)f2.x; a[9]  = (_Float16)f2.y; a[10] = (_Float16)f2.z; a[11] = (_Float16)f2.w;
    a[12] = (_Float16)f3.x; a[13] = (_Float16)f3.y; a[14] = (_Float16)f3.z; a[15] = (_Float16)f3.w;

    // B fragment (32x16 f16): elem j<8 -> K = k0 + hi*8 + j, elem 8+j -> K = k0 + hi*8 + 16 + j, N = n0 + m
    const _Float16* bp = &WldsT[(n0 + m) * WSTRIDE + k0 + hi * 8];  // 16B aligned
    v8h blo = *(const v8h*)(bp);
    v8h bhi = *(const v8h*)(bp + 16);
    v16h b;
#pragma unroll
    for (int j = 0; j < 8; ++j) { b[j] = blo[j]; b[8 + j] = bhi[j]; }

    acc = __builtin_amdgcn_wmma_f32_16x16x32_f16(false, a, false, b,
                                                 (short)0, acc, false, false);
  }

  // D layout: VGPR r -> row (r + hi*8), col (n0 + m)
#pragma unroll
  for (int r = 0; r < 8; ++r)
    Hout[(size_t)(row0 + r + hi * 8) * EMB + n0 + m] = acc[r];
}

// ---------------------------------------------------- sparse aggregate stage
// agg = bias + selfnorm[i]*hW[i]  (self-loop contribution folded in); float4
__global__ void k_agg_init(const float* __restrict__ hW, const float* __restrict__ selfn,
                           const float* __restrict__ bias, float* __restrict__ agg) {
  int gid = blockIdx.x * blockDim.x + threadIdx.x;   // over N_NODES * 16 quads
  if (gid < N_NODES * (EMB / 4)) {
    int i = gid >> 4, q = gid & 15;
    float s = selfn[i];
    float4 h4 = ((const float4*)hW)[gid];
    float4 b4 = ((const float4*)bias)[q];
    float4 o;
    o.x = b4.x + h4.x * s; o.y = b4.y + h4.y * s;
    o.z = b4.z + h4.z * s; o.w = b4.w + h4.w * s;
    ((float4*)agg)[gid] = o;
  }
}

// agg[dst] += norm[e] * hW[src]; one thread per (edge, 4-feature quad):
// float4 gather + 4 f32 atomics, index/norm loads amortized 4x.
__global__ void k_edge_scatter(const int* __restrict__ src, const int* __restrict__ dst,
                               const float* __restrict__ norm, const float* __restrict__ hW,
                               float* __restrict__ agg) {
  int gid = blockIdx.x * blockDim.x + threadIdx.x;   // over N_EDGES * 16 quads
  int e = gid >> 4, q = gid & 15;
  if (e < N_EDGES) {
    float nrm = norm[e];
    float4 v = ((const float4*)(hW + (size_t)src[e] * EMB))[q];
    float* d = agg + (size_t)dst[e] * EMB + q * 4;
    atomicAdd(d + 0, v.x * nrm);
    atomicAdd(d + 1, v.y * nrm);
    atomicAdd(d + 2, v.z * nrm);
    atomicAdd(d + 3, v.w * nrm);
  }
}

__global__ void k_tanh(float* __restrict__ h) {
  int gid = blockIdx.x * blockDim.x + threadIdx.x;
  if (gid < N_NODES * EMB) h[gid] = tanhf(h[gid]);
}

// ------------------------------------------------------------------- pooling
__device__ __forceinline__ void atomicMaxFloat(float* addr, float val) {
  if (val >= 0.0f) atomicMax((int*)addr, __float_as_int(val));
  else             atomicMin((unsigned int*)addr, __float_as_uint(val));
}

// Pool with fused tanh (the 5th layer's activation is applied here, saving a
// full 12.8MB read+write pass over the node embeddings).
__global__ void k_pool_tanh(const float* __restrict__ h, const int* __restrict__ batch,
                            float* __restrict__ gmax, float* __restrict__ gsum,
                            float* __restrict__ counts) {
  int gid = blockIdx.x * blockDim.x + threadIdx.x;
  if (gid < N_NODES * EMB) {
    int i = gid >> 6, f = gid & 63;
    int g = batch[i];
    float v = tanhf(h[gid]);
    atomicMaxFloat(&gmax[(size_t)g * EMB + f], v);
    atomicAdd(&gsum[(size_t)g * EMB + f], v);
    if (f == 0) atomicAdd(&counts[g], 1.0f);
  }
}

// hidden[g] = [gmax | gmean]; out[g] = hidden . W_out + b_out
// d_out layout: [0,512) = out, [512, 512+512*128) = hidden
__global__ __launch_bounds__(128) void k_finalize(const float* __restrict__ gmax,
                                                  const float* __restrict__ gsum,
                                                  const float* __restrict__ counts,
                                                  const float* __restrict__ Wout,
                                                  const float* __restrict__ bout,
                                                  float* __restrict__ out) {
  __shared__ float red[128];
  int g = blockIdx.x, f = threadIdx.x;
  float hv;
  if (f < EMB) {
    hv = gmax[(size_t)g * EMB + f];
  } else {
    float c = counts[g];
    hv = gsum[(size_t)g * EMB + (f - EMB)] / fmaxf(c, 1.0f);
  }
  out[N_GRAPHS + (size_t)g * (2 * EMB) + f] = hv;

  red[f] = hv * Wout[f];
  __syncthreads();
  for (int s = 64; s > 0; s >>= 1) {
    if (f < s) red[f] += red[f + s];
    __syncthreads();
  }
  if (f == 0) out[g] = red[0] + bout[0];
}

// -------------------------------------------------------------------- driver
extern "C" void kernel_launch(void* const* d_in, const int* in_sizes, int n_in,
                              void* d_out, int out_size, void* d_ws, size_t ws_size,
                              hipStream_t stream) {
  (void)in_sizes; (void)n_in; (void)out_size; (void)ws_size;

  const float* x     = (const float*)d_in[0];
  const int*   ei    = (const int*)d_in[1];
  const int*   batch = (const int*)d_in[2];
  const float* W_in  = (const float*)d_in[3];
  const float* b_in  = (const float*)d_in[4];
  const float* W1    = (const float*)d_in[5];  const float* b1 = (const float*)d_in[6];
  const float* W2    = (const float*)d_in[7];  const float* b2 = (const float*)d_in[8];
  const float* W3    = (const float*)d_in[9];  const float* b3 = (const float*)d_in[10];
  const float* W4    = (const float*)d_in[11]; const float* b4 = (const float*)d_in[12];
  const float* W_out = (const float*)d_in[13]; const float* b_out = (const float*)d_in[14];

  const int* src = ei;
  const int* dst = ei + N_EDGES;

  // workspace layout (floats; all offsets multiples of 4 -> float4-safe)
  float* ws     = (float*)d_ws;
  float* dis    = ws;                                   // 50016
  float* selfn  = dis  + 50016;                         // 50016
  float* norm   = selfn + 50016;                        // 800000
  float* hA     = norm + N_EDGES;                       // 3,200,000
  float* hB     = hA + (size_t)N_NODES * EMB;           // 3,200,000
  float* hW     = hB + (size_t)N_NODES * EMB;           // 3,200,000
  float* gmax   = hW + (size_t)N_NODES * EMB;           // 32768
  float* gsum   = gmax + N_GRAPHS * EMB;                // 32768
  float* counts = gsum + N_GRAPHS * EMB;                // 512

  const int TPB = 256;
  const int gNodes  = (N_NODES + TPB - 1) / TPB;
  const int gEdges  = (N_EDGES + TPB - 1) / TPB;
  const int gNodeFt = (N_NODES * EMB + TPB - 1) / TPB;            // 12500
  const int gNodeQ  = (N_NODES * (EMB / 4) + TPB - 1) / TPB;      // 3125
  const int gEdgeQ  = (N_EDGES * (EMB / 4) + TPB - 1) / TPB;      // 50000

  // ---- GCN normalization
  k_fill<<<gNodes, TPB, 0, stream>>>(dis, N_NODES, 1.0f);
  k_deg_edges<<<gEdges, TPB, 0, stream>>>(dst, dis);
  k_deg_fin<<<gNodes, TPB, 0, stream>>>(dis, selfn);
  k_edge_norm<<<gEdges, TPB, 0, stream>>>(src, dst, dis, norm);

  // ---- 5 GCN layers (ping-pong hA/hB); tanh of last layer fused into pooling
  auto layer64 = [&](const float* hin, const float* W, const float* b,
                     float* hout, bool applyTanh) {
    gcn_matmul_wmma<EMB><<<N_NODES / 16, 128, 0, stream>>>(hin, W, hW);
    k_agg_init<<<gNodeQ, TPB, 0, stream>>>(hW, selfn, b, hout);
    k_edge_scatter<<<gEdgeQ, TPB, 0, stream>>>(src, dst, norm, hW, hout);
    if (applyTanh) k_tanh<<<gNodeFt, TPB, 0, stream>>>(hout);
  };

  // layer 0 (K = 32)
  gcn_matmul_wmma<N_FEAT><<<N_NODES / 16, 128, 0, stream>>>(x, W_in, hW);
  k_agg_init<<<gNodeQ, TPB, 0, stream>>>(hW, selfn, b_in, hA);
  k_edge_scatter<<<gEdgeQ, TPB, 0, stream>>>(src, dst, norm, hW, hA);
  k_tanh<<<gNodeFt, TPB, 0, stream>>>(hA);

  layer64(hA, W1, b1, hB, true);
  layer64(hB, W2, b2, hA, true);
  layer64(hA, W3, b3, hB, true);
  layer64(hB, W4, b4, hA, false);   // pre-tanh embeddings in hA

  // ---- pooling (tanh fused) + readout
  k_fill<<<(N_GRAPHS * EMB + TPB - 1) / TPB, TPB, 0, stream>>>(gmax, N_GRAPHS * EMB, -INFINITY);
  k_fill<<<(N_GRAPHS * EMB + N_GRAPHS + TPB - 1) / TPB, TPB, 0, stream>>>(gsum, N_GRAPHS * EMB + N_GRAPHS, 0.0f);
  k_pool_tanh<<<gNodeFt, TPB, 0, stream>>>(hA, batch, gmax, gsum, counts);
  k_finalize<<<N_GRAPHS, 128, 0, stream>>>(gmax, gsum, counts, W_out, b_out, (float*)d_out);
}
